// MultiScaleTimeMixer_49194555408699
// MI455X (gfx1250) — compile-verified, matrix operand order checked by reading, not run on hardware
//
#include <hip/hip_runtime.h>
#include <cstdint>

// ---------------------------------------------------------------------------
// MultiScaleTimeMixer for MI455X (gfx1250, wave32, WMMA).
// All matrix work goes through v_wmma_f32_16x16x32_f16 (f16 in, f32 accum).
// Double-buffered LDS GEMM, one barrier per K-step, 8 WMMA per wave per step,
// vectorized (b128) epilogue stores.
// ---------------------------------------------------------------------------

#define B_SZ   64
#define T_LEN  512
#define C_LEN  512
#define T2_LEN 256

typedef __attribute__((ext_vector_type(16))) _Float16 v16h;
typedef __attribute__((ext_vector_type(8)))  _Float16 v8h;
typedef __attribute__((ext_vector_type(8)))  float    v8f;

// ---------------------------------------------------------------------------
// LayerNorm stats: one block per batch, reduce 262144 elements (sum, sumsq).
// ---------------------------------------------------------------------------
__global__ void msm_ln_stats_kernel(const float* __restrict__ x,
                                    float* __restrict__ stats) {
    __shared__ float s1[256];
    __shared__ float s2[256];
    const int b = blockIdx.x;
    const float* xb = x + (size_t)b * (T_LEN * C_LEN);
    float sum = 0.f, sq = 0.f;
    for (int i = threadIdx.x; i < T_LEN * C_LEN; i += 256) {
        float v = xb[i];
        sum += v;
        sq  += v * v;
    }
    s1[threadIdx.x] = sum;
    s2[threadIdx.x] = sq;
    __syncthreads();
    for (int s = 128; s > 0; s >>= 1) {
        if (threadIdx.x < s) {
            s1[threadIdx.x] += s1[threadIdx.x + s];
            s2[threadIdx.x] += s2[threadIdx.x + s];
        }
        __syncthreads();
    }
    if (threadIdx.x == 0) {
        const float n = (float)(T_LEN * C_LEN);
        float mean = s1[0] / n;
        float var  = s2[0] / n - mean * mean;
        stats[b * 2 + 0] = mean;
        stats[b * 2 + 1] = rsqrtf(var + 1e-5f);
    }
}

// ---------------------------------------------------------------------------
// LN apply + affine + transpose [B,T,C] -> [B,C,T], emit f16.
// gamma/beta are [T,C] broadcast against xp[B,C,T]: element (b,c,t) uses
// gamma[c*T + t] (T==C==512 makes the broadcast purely positional).
// ---------------------------------------------------------------------------
__global__ void msm_ln_apply_transpose_kernel(const float* __restrict__ x,
                                              const float* __restrict__ gamma,
                                              const float* __restrict__ beta,
                                              const float* __restrict__ stats,
                                              _Float16* __restrict__ xn) {
    __shared__ _Float16 tile[32][33];
    const int b  = blockIdx.z;
    const int t0 = blockIdx.x * 32;
    const int c0 = blockIdx.y * 32;
    const float mean = stats[b * 2 + 0];
    const float inv  = stats[b * 2 + 1];
    const int tx = threadIdx.x;   // 0..31
    const int ty = threadIdx.y;   // 0..7
#pragma unroll
    for (int i = 0; i < 4; ++i) {
        int t = t0 + ty + 8 * i;
        int c = c0 + tx;
        float v  = x[((size_t)b * T_LEN + t) * C_LEN + c];
        float g  = gamma[(size_t)c * T_LEN + t];
        float be = beta [(size_t)c * T_LEN + t];
        tile[ty + 8 * i][tx] = (_Float16)((v - mean) * inv * g + be);
    }
    __syncthreads();
#pragma unroll
    for (int i = 0; i < 4; ++i) {
        int c = c0 + ty + 8 * i;
        int t = t0 + tx;
        xn[((size_t)b * C_LEN + c) * T_LEN + t] = tile[tx][ty + 8 * i];
    }
}

// ---------------------------------------------------------------------------
// Vectorized f32 -> f16 copy (8 elements / thread, b128 in & out).
// ---------------------------------------------------------------------------
__global__ void msm_f32_to_f16_kernel(const float* __restrict__ src,
                                      _Float16* __restrict__ dst, int n8) {
    int i = blockIdx.x * 256 + threadIdx.x;
    if (i >= n8) return;
    const float4* s = (const float4*)src + 2 * (size_t)i;
    float4 a = s[0];
    float4 b = s[1];
    v8h h;
    h[0] = (_Float16)a.x; h[1] = (_Float16)a.y;
    h[2] = (_Float16)a.z; h[3] = (_Float16)a.w;
    h[4] = (_Float16)b.x; h[5] = (_Float16)b.y;
    h[6] = (_Float16)b.z; h[7] = (_Float16)b.w;
    ((v8h*)dst)[i] = h;
}

// ---------------------------------------------------------------------------
// Lower-triangular mask + f16 convert for the TriU weights W[i,j], j<=i kept.
// ---------------------------------------------------------------------------
__global__ void msm_mask_to_f16_kernel(const float* __restrict__ w,
                                       _Float16* __restrict__ dst, int n) {
    int idx = blockIdx.x * 256 + threadIdx.x;
    if (idx >= n * n) return;
    int i = idx / n;
    int j = idx - i * n;
    float v = (j <= i) ? w[idx] : 0.f;
    dst[idx] = (_Float16)v;
}

// ---------------------------------------------------------------------------
// conv_w [O=C, I=C, H=2] -> Wc2 [O, 2C] with Wc2[o, di*C + i] = conv_w[o,i,di]
// so that conv == GEMM against x viewed as [B*T2, 2C].
// ---------------------------------------------------------------------------
__global__ void msm_convw_rearrange_kernel(const float* __restrict__ w,
                                           _Float16* __restrict__ dst) {
    int idx = blockIdx.x * 256 + threadIdx.x;   // < 512*1024
    int o  = idx >> 10;
    int r  = idx & 1023;
    int di = r >> 9;
    int i  = r & 511;
    dst[idx] = (_Float16)w[(size_t)o * 1024 + i * 2 + di];
}

// ---------------------------------------------------------------------------
// WMMA GEMM: D[M,N] = A[M,K] * Bw[N,K]^T + bias[N] (+hardswish)
// Block = 256 threads (8 waves), tile 128x128, K-step 32, double-buffered LDS.
// Waves in 2(M) x 4(N) grid; each wave owns 64x32 = 4x2 fragments = 8 WMMA
// per K-step, 1 barrier per K-step.
// STORE_MODE: 0 = f32, row stride ldc, column offset col_off (concat output)
//             1 = f16 row-major, stride N (LDS-staged b128 stores)
//             2 = f16 transposed conv store: row=(b,t'), col=o ->
//                 out[(b*C + o)*T2 + t'] (8 contiguous t' per lane -> b128)
// CAUSAL: weights are tril-masked, so K loop can stop at n0+BN.
// ---------------------------------------------------------------------------
template <int STORE_MODE, bool HSWISH, bool CAUSAL>
__global__ __launch_bounds__(256, 2)
void msm_gemm_wmma_kernel(const _Float16* __restrict__ A,
                          const _Float16* __restrict__ Bw,
                          const float* __restrict__ bias,
                          void* __restrict__ out,
                          int M, int N, int K, int ldc, int col_off) {
    constexpr int BM = 128, BN = 128, BK = 32;
    constexpr int LDSS = 40;          // 40-half pitch: conflict-free b128
    constexpr int BUFB = (BM + BN) * LDSS * 2;   // bytes per double-buffer slot
    __shared__ __align__(16) char smem[2 * BUFB];

    const int tid  = threadIdx.x;
    const int lane = tid & 31;
    const int wid  = tid >> 5;     // 0..7
    const int wm   = wid >> 2;     // 0..1 : M sub-tile (64 rows)
    const int wn   = wid & 3;      // 0..3 : N sub-tile (32 cols)
    const int m0 = blockIdx.x * BM;
    const int n0 = blockIdx.y * BN;

    const int kend = CAUSAL ? ((n0 + BN) < K ? (n0 + BN) : K) : K;
    const int nk   = kend / BK;

    v8f acc[4][2] = {};

    const int lm     = lane & 15;            // fragment row (M for A, N for B)
    const int lkbase = (lane >> 4) * 8;      // K base column per lane-half

    // global staging: each thread moves 2x16B for A and 2x16B for B per step
    const int srow = tid >> 2;               // 0..63
    const int scol = (tid & 3) * 8;          // 0,8,16,24
    const _Float16* gA0 = A  + (size_t)(m0 + srow) * K + scol;
    const _Float16* gA1 = A  + (size_t)(m0 + srow + 64) * K + scol;
    const _Float16* gB0 = Bw + (size_t)(n0 + srow) * K + scol;
    const _Float16* gB1 = Bw + (size_t)(n0 + srow + 64) * K + scol;

    v8h ra0, ra1, rb0, rb1;
    auto load_tiles = [&](int k0) {
        ra0 = *(const v8h*)(gA0 + k0);
        ra1 = *(const v8h*)(gA1 + k0);
        rb0 = *(const v8h*)(gB0 + k0);
        rb1 = *(const v8h*)(gB1 + k0);
    };
    auto store_tiles = [&](int d) {
        _Float16* sA = (_Float16*)(smem + d * BUFB);
        _Float16* sB = sA + BM * LDSS;
        *(v8h*)(sA + srow * LDSS + scol)        = ra0;
        *(v8h*)(sA + (srow + 64) * LDSS + scol) = ra1;
        *(v8h*)(sB + srow * LDSS + scol)        = rb0;
        *(v8h*)(sB + (srow + 64) * LDSS + scol) = rb1;
    };

    load_tiles(0);
    store_tiles(0);
    __syncthreads();

    for (int kt = 0; kt < nk; ++kt) {
        const int cur = kt & 1;
        const bool has_next = (kt + 1 < nk);
        if (has_next) load_tiles((kt + 1) * BK);

        _Float16* sA = (_Float16*)(smem + cur * BUFB);
        _Float16* sB = sA + BM * LDSS;

        v16h af[4], bf[2];
#pragma unroll
        for (int mi = 0; mi < 4; ++mi) {
            const _Float16* p = sA + (wm * 64 + mi * 16 + lm) * LDSS + lkbase;
            v8h lo = *(const v8h*)(p);
            v8h hi = *(const v8h*)(p + 16);
            af[mi] = __builtin_shufflevector(lo, hi, 0, 1, 2, 3, 4, 5, 6, 7,
                                             8, 9, 10, 11, 12, 13, 14, 15);
        }
#pragma unroll
        for (int ni = 0; ni < 2; ++ni) {
            const _Float16* p = sB + (wn * 32 + ni * 16 + lm) * LDSS + lkbase;
            v8h lo = *(const v8h*)(p);
            v8h hi = *(const v8h*)(p + 16);
            bf[ni] = __builtin_shufflevector(lo, hi, 0, 1, 2, 3, 4, 5, 6, 7,
                                             8, 9, 10, 11, 12, 13, 14, 15);
        }
        if (has_next) store_tiles(1 - cur);
        __syncthreads();

#pragma unroll
        for (int mi = 0; mi < 4; ++mi)
#pragma unroll
            for (int ni = 0; ni < 2; ++ni)
                acc[mi][ni] = __builtin_amdgcn_wmma_f32_16x16x32_f16(
                    false, af[mi], false, bf[ni], (short)0, acc[mi][ni],
                    false, false);
    }
    __syncthreads();

    // ---- epilogue -------------------------------------------------------
    // D fragment layout: lane holds col n = lane&15, rows r + 8*(lane>=16).
    const int nlane = lane & 15;
    const int mo    = (lane >> 4) * 8;
    // staging view (modes 0/1): per-wave 16x20 f32 region in smem
    float* stg = (float*)smem + (size_t)wid * 320;
    const int orow = lane >> 1;           // 0..15 : output row within frag
    const int ocol = (lane & 1) * 8;      // 0 or 8 : output col chunk

#pragma unroll
    for (int mi = 0; mi < 4; ++mi) {
#pragma unroll
        for (int ni = 0; ni < 2; ++ni) {
            const int grow0 = m0 + wm * 64 + mi * 16;
            const int gcol0 = n0 + wn * 32 + ni * 16;
            const float bv = bias[gcol0 + nlane];
            if constexpr (STORE_MODE == 2) {
                // 8 consecutive t' per lane -> single b128 store
                v8h h;
#pragma unroll
                for (int r = 0; r < 8; ++r)
                    h[r] = (_Float16)(acc[mi][ni][r] + bv);
                int grow = grow0 + mo;
                int bb = grow >> 8;          // T2 = 256
                int tt = grow & 255;
                *(v8h*)((_Float16*)out +
                        ((size_t)bb * C_LEN + gcol0 + nlane) * T2_LEN + tt) = h;
            } else {
                // stage fragment (f32) through LDS to get b128 row stores
#pragma unroll
                for (int r = 0; r < 8; ++r) {
                    float v = acc[mi][ni][r] + bv;
                    if (HSWISH)
                        v = v * fminf(fmaxf(v + 3.f, 0.f), 6.f) * (1.f / 6.f);
                    stg[(mo + r) * 20 + nlane] = v;
                }
                __syncthreads();
                const float* s = stg + orow * 20 + ocol;
                if constexpr (STORE_MODE == 0) {
                    float* dst = (float*)out + (size_t)(grow0 + orow) * ldc +
                                 col_off + gcol0 + ocol;
                    *(float4*)(dst)     = *(const float4*)(s);
                    *(float4*)(dst + 4) = *(const float4*)(s + 4);
                } else {
                    v8h h;
#pragma unroll
                    for (int j = 0; j < 8; ++j) h[j] = (_Float16)s[j];
                    *(v8h*)((_Float16*)out +
                            (size_t)(grow0 + orow) * N + gcol0 + ocol) = h;
                }
                __syncthreads();
            }
        }
    }
}

// ---------------------------------------------------------------------------
// Workspace layout (bytes, 256-aligned). Total ~136.6 MB.
// ---------------------------------------------------------------------------
static constexpr size_t OFF_STATS = 0;                        // 64*2 f32
static constexpr size_t OFF_XN    = 512;                      // [B*C, T] f16
static constexpr size_t OFF_XH    = OFF_XN + 33554432;        // [B*T2, 2C] f16
static constexpr size_t OFF_Y1    = OFF_XH + 33554432;        // [B*C, T] f16
static constexpr size_t OFF_ZC    = OFF_Y1 + 33554432;        // [B*C, T2] f16
static constexpr size_t OFF_Z3    = OFF_ZC + 16777216;        // [B*C, T2] f16
static constexpr size_t OFF_W1    = OFF_Z3 + 16777216;        // 512x512 f16
static constexpr size_t OFF_W2    = OFF_W1 + 524288;
static constexpr size_t OFF_W3    = OFF_W2 + 524288;          // 256x256 f16
static constexpr size_t OFF_W4    = OFF_W3 + 131072;
static constexpr size_t OFF_WC    = OFF_W4 + 131072;          // 512x1024 f16

extern "C" void kernel_launch(void* const* d_in, const int* in_sizes, int n_in,
                              void* d_out, int out_size, void* d_ws,
                              size_t ws_size, hipStream_t stream) {
    (void)in_sizes; (void)n_in; (void)out_size; (void)ws_size;

    const float* x        = (const float*)d_in[0];
    const float* ln_gamma = (const float*)d_in[1];
    const float* ln_beta  = (const float*)d_in[2];
    const float* w1       = (const float*)d_in[3];
    const float* b1       = (const float*)d_in[4];
    const float* w2       = (const float*)d_in[5];
    const float* b2       = (const float*)d_in[6];
    const float* conv_w   = (const float*)d_in[7];
    const float* conv_b   = (const float*)d_in[8];
    const float* w3       = (const float*)d_in[9];
    const float* b3       = (const float*)d_in[10];
    const float* w4       = (const float*)d_in[11];
    const float* b4       = (const float*)d_in[12];
    float* out = (float*)d_out;

    char* ws = (char*)d_ws;
    float*    stats = (float*)(ws + OFF_STATS);
    _Float16* xn_h  = (_Float16*)(ws + OFF_XN);
    _Float16* x_h   = (_Float16*)(ws + OFF_XH);
    _Float16* y1_h  = (_Float16*)(ws + OFF_Y1);
    _Float16* zc_h  = (_Float16*)(ws + OFF_ZC);
    _Float16* z3_h  = (_Float16*)(ws + OFF_Z3);
    _Float16* w1m   = (_Float16*)(ws + OFF_W1);
    _Float16* w2m   = (_Float16*)(ws + OFF_W2);
    _Float16* w3m   = (_Float16*)(ws + OFF_W3);
    _Float16* w4m   = (_Float16*)(ws + OFF_W4);
    _Float16* wc_h  = (_Float16*)(ws + OFF_WC);

    // --- prep: stats, LN+transpose, f16 conversions, weight rearranges -----
    msm_ln_stats_kernel<<<B_SZ, 256, 0, stream>>>(x, stats);
    msm_ln_apply_transpose_kernel<<<dim3(16, 16, 64), dim3(32, 8), 0, stream>>>(
        x, ln_gamma, ln_beta, stats, xn_h);
    msm_f32_to_f16_kernel<<<(B_SZ * T_LEN * C_LEN / 8) / 256, 256, 0, stream>>>(
        x, x_h, B_SZ * T_LEN * C_LEN / 8);
    msm_mask_to_f16_kernel<<<(512 * 512) / 256, 256, 0, stream>>>(w1, w1m, 512);
    msm_mask_to_f16_kernel<<<(512 * 512) / 256, 256, 0, stream>>>(w2, w2m, 512);
    msm_mask_to_f16_kernel<<<(256 * 256) / 256, 256, 0, stream>>>(w3, w3m, 256);
    msm_mask_to_f16_kernel<<<(256 * 256) / 256, 256, 0, stream>>>(w4, w4m, 256);
    msm_convw_rearrange_kernel<<<(512 * 1024) / 256, 256, 0, stream>>>(conv_w,
                                                                       wc_h);

    // --- scale 0: triu1 (+hardswish) then triu2 -> out cols [0,512) --------
    msm_gemm_wmma_kernel<1, true, true>
        <<<dim3(256, 4), 256, 0, stream>>>(xn_h, w1m, b1, y1_h,
                                           32768, 512, 512, 512, 0);
    msm_gemm_wmma_kernel<0, false, true>
        <<<dim3(256, 4), 256, 0, stream>>>(y1_h, w2m, b2, out,
                                           32768, 512, 512, 768, 0);

    // --- scale 1: conv-as-GEMM (transposed f16 store) ----------------------
    msm_gemm_wmma_kernel<2, false, false>
        <<<dim3(128, 4), 256, 0, stream>>>(x_h, wc_h, conv_b, zc_h,
                                           16384, 512, 1024, 0, 0);
    // triu3 (+hardswish), then triu4 -> out cols [512,768) ------------------
    msm_gemm_wmma_kernel<1, true, true>
        <<<dim3(256, 2), 256, 0, stream>>>(zc_h, w3m, b3, z3_h,
                                           32768, 256, 256, 256, 0);
    msm_gemm_wmma_kernel<0, false, true>
        <<<dim3(256, 2), 256, 0, stream>>>(z3_h, w4m, b4, out,
                                           32768, 256, 256, 768, 512);
}
